// S5Layer_23175643529601
// MI455X (gfx1250) — matrix operand order repaired and checked by reading, not compiled, not run on hardware
//
#include <hip/hip_runtime.h>
#include <hip/hip_bf16.h>
#include <stdint.h>

// ---------------------------------------------------------------------------
// S5 layer for MI455X (gfx1250): bf16 WMMA GEMMs + chunked f32 scan.
//   B=8, S=4096, F=768, N=128.  M = B*S = 32768 rows.
//   GEMM1:  Bu[M,256] = u[M,768] @ W1[768,256]      (W1 = [B_bar_re; B_bar_im]ᵀ)
//   scan :  x[s] = A_bar * x[s-1] + Bu[s]  (complex, per n), 3-pass chunked
//   GEMM2:  y[M,768]  = X[M,256] @ W2[256,768] + D⊙u (W2 = [C_reᵀ; -C_imᵀ])
// CDNA5 paths: v_wmma_f32_16x16x32_bf16, global_load_async_to_lds_b128,
// global_store_async_from_lds_b128, s_wait_asynccnt; double-buffered LDS.
// ---------------------------------------------------------------------------

#define SDIM   4096
#define BDIM   8
#define FDIM   768
#define NDIM   128
#define MROWS  (BDIM * SDIM)      // 32768
#define NC1    (2 * NDIM)         // 256 (re | im)
#define LCHUNK 64
#define NCHUNK (SDIM / LCHUNK)    // 64

typedef __attribute__((ext_vector_type(16))) __bf16 v16bf;
typedef __attribute__((ext_vector_type(8)))  float  v8f;

__device__ __forceinline__ unsigned short f2bf(float f) {
  __bf16 h = (__bf16)f;
  return *(unsigned short*)&h;
}

__device__ __forceinline__ v8f wmma_bf16(v16bf a, v16bf b, v8f c) {
  return __builtin_amdgcn_wmma_f32_16x16x32_bf16(
      /*neg_a=*/false, a, /*neg_b=*/false, b,
      /*c_mod=*/(short)0, c, /*reuse_a=*/false, /*reuse_b=*/false);
}

// ---- CDNA5 async LDS <-> global (ASYNCcnt) -------------------------------
// Low 32 bits of a flat shared pointer == wave-relative LDS byte address.
__device__ __forceinline__ void async_load_b128(void* lds_ptr, const void* gptr) {
  unsigned lds_off = (unsigned)(uintptr_t)lds_ptr;
  asm volatile("global_load_async_to_lds_b128 %0, %1, off"
               :: "v"(lds_off), "v"(gptr) : "memory");
}
__device__ __forceinline__ void async_store_b128(void* gptr, const void* lds_ptr) {
  unsigned lds_off = (unsigned)(uintptr_t)lds_ptr;
  asm volatile("global_store_async_from_lds_b128 %0, %1, off"
               :: "v"(gptr), "v"(lds_off) : "memory");
}
__device__ __forceinline__ void wait_asynccnt0() {
  asm volatile("s_wait_asynccnt 0" ::: "memory");
}

// A fragment: 16x32 tile, row-major in LDS with stride lda (elements).
// lanes 0-15 row M=lane, K 0-7 (v0-3) & 16-23 (v4-7);
// lanes 16-31 row M=lane-16, K 8-15 & 24-31. Two bf16 per VGPR.
__device__ __forceinline__ v16bf load_a_frag(const __bf16* As, int lda, int lane) {
  const int m = lane & 15, h = lane >> 4;
  const __bf16* row = As + m * lda;
  union { v16bf v; unsigned u[8]; } a;
#pragma unroll
  for (int g = 0; g < 8; ++g) {
    int k0 = ((g >> 2) << 4) + (h << 3) + ((g & 3) << 1);
    a.u[g] = *(const unsigned*)(row + k0);
  }
  return a.v;
}

// B fragment: 32x16 tile (K x N), row-major in LDS with stride ldb.
// lanes 0-15 col N=lane hold K 0-7 & 16-23 (2/VGPR); lanes 16-31 K 8-15 & 24-31.
__device__ __forceinline__ v16bf load_b_frag(const __bf16* Bs, int ldb, int n_off, int lane) {
  const int n = (lane & 15) + n_off, h = lane >> 4;
  union { v16bf v; unsigned u[8]; } b;
#pragma unroll
  for (int g = 0; g < 8; ++g) {
    int k0 = ((g >> 2) << 4) + (h << 3) + ((g & 3) << 1);
    unsigned lo = *(const unsigned short*)(Bs + (size_t)k0 * ldb + n);
    unsigned hi = *(const unsigned short*)(Bs + (size_t)(k0 + 1) * ldb + n);
    b.u[g] = lo | (hi << 16);
  }
  return b.v;
}

// ---------------------------------------------------------------------------
// Param prep
// ---------------------------------------------------------------------------
__global__ void prep_params_kernel(const float* __restrict__ log_A_real,
                                   const float* __restrict__ A_imag,
                                   const float* __restrict__ log_dt,
                                   float2* __restrict__ A_bar,
                                   float2* __restrict__ Apow) {
  int n = threadIdx.x;
  if (n >= NDIM) return;
  float dt  = expf(log_dt[n]);
  float ar  = -expf(log_A_real[n]);
  float ai  = A_imag[n];
  float mag = expf(ar * dt);
  float Abr = mag * cosf(ai * dt);
  float Abi = mag * sinf(ai * dt);
  A_bar[n] = make_float2(Abr, Abi);
  float pr = 1.f, pi = 0.f;
  for (int k = 0; k <= LCHUNK; ++k) {
    Apow[n * (LCHUNK + 1) + k] = make_float2(pr, pi);
    float nr = pr * Abr - pi * Abi;
    float ni = pr * Abi + pi * Abr;
    pr = nr; pi = ni;
  }
}

__global__ __launch_bounds__(256) void prep_w1_kernel(const float* __restrict__ B_re,
                                                      const float* __restrict__ B_im,
                                                      const float* __restrict__ log_A_real,
                                                      const float* __restrict__ A_imag,
                                                      const float2* __restrict__ A_bar,
                                                      __bf16* __restrict__ W1) {
  int idx = blockIdx.x * 256 + threadIdx.x;   // < N*F
  int f = idx >> 7, n = idx & 127;
  float2 Ab = A_bar[n];
  float adr = -expf(log_A_real[n]) + 1e-8f;
  float adi = A_imag[n];
  float den = adr * adr + adi * adi;
  float gr = Ab.x - 1.0f, gi = Ab.y;
  float fr = (gr * adr + gi * adi) / den;
  float fi = (gi * adr - gr * adi) / den;
  float br = B_re[(size_t)n * FDIM + f];
  float bi = B_im[(size_t)n * FDIM + f];
  W1[(size_t)f * NC1 + n]        = (__bf16)(fr * br - fi * bi);
  W1[(size_t)f * NC1 + NDIM + n] = (__bf16)(fr * bi + fi * br);
}

__global__ __launch_bounds__(256) void prep_w2_kernel(const float* __restrict__ C_re,
                                                      const float* __restrict__ C_im,
                                                      __bf16* __restrict__ W2) {
  int idx = blockIdx.x * 256 + threadIdx.x;   // < N*F
  int n = idx / FDIM, f = idx % FDIM;
  W2[(size_t)n * FDIM + f]          = (__bf16)( C_re[(size_t)f * NDIM + n]);
  W2[(size_t)(NDIM + n) * FDIM + f] = (__bf16)(-C_im[(size_t)f * NDIM + n]);
}

// ---------------------------------------------------------------------------
// GEMM1: Bu[M,256] = bf16(u[M,768]) @ W1[768,256].
// Double-buffered: B tile via async LDS loads; A tile (f32->bf16) prefetched
// into VGPRs before the WMMA block, converted/stored after it.
// Block tile 128x64, 8 waves (4x2), wave tile 32x32.
// ---------------------------------------------------------------------------
__global__ __launch_bounds__(256) void gemm1_kernel(const float* __restrict__ u,
                                                    const __bf16* __restrict__ W1,
                                                    float* __restrict__ Bu) {
  constexpr int BM = 128, BN = 64, BK = 32, LDA = 36, LDB = 72;
  __shared__ __align__(16) __bf16 As[2][BM * LDA];   // 2 x 9 KB
  __shared__ __align__(16) __bf16 Bs[2][BK * LDB];   // 2 x 4.5 KB
  const int tid = threadIdx.x, lane = tid & 31, wid = tid >> 5;
  const int wm = wid >> 1, wn = wid & 1;
  const int row0 = blockIdx.x * BM, col0 = blockIdx.y * BN;
  const int br_ = tid >> 3, bc8 = tid & 7;           // B tile: one b128 / thread
  v8f acc[2][2] = {};

  // prologue: stage tile 0
  async_load_b128(&Bs[0][br_ * LDB + bc8 * 8],
                  W1 + (size_t)br_ * NC1 + col0 + bc8 * 8);
#pragma unroll
  for (int i = 0; i < 4; ++i) {
    int slot = tid + i * 256;                        // 1024 float4 slots
    int r = slot >> 3, c4 = slot & 7;
    float4 v = *(const float4*)(u + (size_t)(row0 + r) * FDIM + c4 * 4);
    unsigned lo = f2bf(v.x) | ((unsigned)f2bf(v.y) << 16);
    unsigned hi = f2bf(v.z) | ((unsigned)f2bf(v.w) << 16);
    *(uint2*)(&As[0][r * LDA + c4 * 4]) = make_uint2(lo, hi);
  }
  wait_asynccnt0();
  __syncthreads();

  int cur = 0;
  for (int k0 = 0; k0 < FDIM; k0 += BK, cur ^= 1) {
    const int nxt = cur ^ 1;
    const bool has_next = (k0 + BK) < FDIM;
    float4 pf[4];
    if (has_next) {
      async_load_b128(&Bs[nxt][br_ * LDB + bc8 * 8],
                      W1 + (size_t)(k0 + BK + br_) * NC1 + col0 + bc8 * 8);
#pragma unroll
      for (int i = 0; i < 4; ++i) {
        int slot = tid + i * 256;
        int r = slot >> 3, c4 = slot & 7;
        pf[i] = *(const float4*)(u + (size_t)(row0 + r) * FDIM + (k0 + BK) + c4 * 4);
      }
    }
    v16bf a0 = load_a_frag(&As[cur][(wm * 32) * LDA], LDA, lane);
    v16bf a1 = load_a_frag(&As[cur][(wm * 32 + 16) * LDA], LDA, lane);
    v16bf b0 = load_b_frag(&Bs[cur][0], LDB, wn * 32, lane);
    v16bf b1 = load_b_frag(&Bs[cur][0], LDB, wn * 32 + 16, lane);
    acc[0][0] = wmma_bf16(a0, b0, acc[0][0]);
    acc[0][1] = wmma_bf16(a0, b1, acc[0][1]);
    acc[1][0] = wmma_bf16(a1, b0, acc[1][0]);
    acc[1][1] = wmma_bf16(a1, b1, acc[1][1]);
    if (has_next) {
#pragma unroll
      for (int i = 0; i < 4; ++i) {
        int slot = tid + i * 256;
        int r = slot >> 3, c4 = slot & 7;
        unsigned lo = f2bf(pf[i].x) | ((unsigned)f2bf(pf[i].y) << 16);
        unsigned hi = f2bf(pf[i].z) | ((unsigned)f2bf(pf[i].w) << 16);
        *(uint2*)(&As[nxt][r * LDA + c4 * 4]) = make_uint2(lo, hi);
      }
    }
    wait_asynccnt0();
    __syncthreads();
  }
  const int h = lane >> 4, cn = lane & 15;
#pragma unroll
  for (int i = 0; i < 2; ++i)
#pragma unroll
    for (int j = 0; j < 2; ++j)
#pragma unroll
      for (int r = 0; r < 8; ++r)
        Bu[(size_t)(row0 + wm * 32 + i * 16 + h * 8 + r) * NC1 +
           col0 + wn * 32 + j * 16 + cn] = acc[i][j][r];
}

// ---------------------------------------------------------------------------
// Scan pass 1: chunk-local scan; 64 KB chunk staged with async LDS DMA.
// ---------------------------------------------------------------------------
__global__ __launch_bounds__(256) void scan_chunk_kernel(float* __restrict__ Bu,
                                                         const float2* __restrict__ A_bar,
                                                         float2* __restrict__ carry) {
  __shared__ __align__(16) float ch[LCHUNK * NC1];   // 64 KB
  const int c = blockIdx.x, b = blockIdx.y;
  const size_t base = ((size_t)b * SDIM + (size_t)c * LCHUNK) * NC1;
  const int tid = threadIdx.x;
#pragma unroll
  for (int i = 0; i < 16; ++i) {
    int slot = tid + i * 256;                        // 4096 b128 slots
    async_load_b128(ch + slot * 4, Bu + base + (size_t)slot * 4);
  }
  wait_asynccnt0();
  __syncthreads();
  if (tid < NDIM) {
    const float2 A = A_bar[tid];
    float hr = 0.f, hi = 0.f;
#pragma unroll 4
    for (int k = 0; k < LCHUNK; ++k) {
      float br = ch[k * NC1 + tid];
      float bi = ch[k * NC1 + NDIM + tid];
      float nr = A.x * hr - A.y * hi + br;
      float ni = A.x * hi + A.y * hr + bi;
      hr = nr; hi = ni;
      ch[k * NC1 + tid] = hr;
      ch[k * NC1 + NDIM + tid] = hi;
    }
    carry[((size_t)b * NCHUNK + c) * NDIM + tid] = make_float2(hr, hi);
  }
  __syncthreads();
#pragma unroll
  for (int i = 0; i < 16; ++i) {
    int slot = tid + i * 256;
    async_store_b128(Bu + base + (size_t)slot * 4, ch + slot * 4);
  }
  wait_asynccnt0();
}

// Scan pass 2: scan carries across 64 chunks. h' = A^L * h + carry.
__global__ void carry_scan_kernel(const float2* __restrict__ carry,
                                  const float2* __restrict__ Apow,
                                  float2* __restrict__ hin) {
  int n = threadIdx.x, b = blockIdx.x;
  if (n >= NDIM) return;
  float2 AL = Apow[n * (LCHUNK + 1) + LCHUNK];
  float hr = 0.f, hi = 0.f;
  for (int c = 0; c < NCHUNK; ++c) {
    size_t idx = ((size_t)b * NCHUNK + c) * NDIM + n;
    hin[idx] = make_float2(hr, hi);
    float2 cr = carry[idx];
    float nr = AL.x * hr - AL.y * hi + cr.x;
    float ni = AL.x * hi + AL.y * hr + cr.y;
    hr = nr; hi = ni;
  }
}

// Scan pass 3: x = A^(k+1)*h_in + local; pack to bf16 for GEMM2.
__global__ __launch_bounds__(256) void fixup_kernel(const float* __restrict__ Bu,
                                                    const float2* __restrict__ hin,
                                                    const float2* __restrict__ Apow,
                                                    __bf16* __restrict__ X) {
  const int c = blockIdx.x, b = blockIdx.y;
  const size_t row0 = (size_t)b * SDIM + (size_t)c * LCHUNK;
  const int tid = threadIdx.x;
  const int n = tid & 127, kbase = tid >> 7;
  const float2 hh = hin[((size_t)b * NCHUNK + c) * NDIM + n];
#pragma unroll 4
  for (int i = 0; i < 32; ++i) {
    int k = kbase + 2 * i;
    float2 p = Apow[n * (LCHUNK + 1) + k + 1];
    size_t base = (row0 + k) * NC1;
    float xr = Bu[base + n]        + p.x * hh.x - p.y * hh.y;
    float xi = Bu[base + NDIM + n] + p.x * hh.y + p.y * hh.x;
    X[base + n]        = (__bf16)xr;
    X[base + NDIM + n] = (__bf16)xi;
  }
}

// ---------------------------------------------------------------------------
// GEMM2: y[M,768] = X[M,256] @ W2[256,768] + D⊙u.
// Fully async-staged, double-buffered. Block tile 128x128, wave tile 32x64.
// ---------------------------------------------------------------------------
__global__ __launch_bounds__(256) void gemm2_kernel(const __bf16* __restrict__ X,
                                                    const __bf16* __restrict__ W2,
                                                    const float* __restrict__ u,
                                                    const float* __restrict__ Dvec,
                                                    float* __restrict__ y) {
  constexpr int BM = 128, BN = 128, BK = 32, LDA = 40, LDB = 136;
  __shared__ __align__(16) __bf16 As[2][BM * LDA];   // 2 x 10 KB
  __shared__ __align__(16) __bf16 Bs[2][BK * LDB];   // 2 x 8.5 KB
  const int tid = threadIdx.x, lane = tid & 31, wid = tid >> 5;
  const int wm = wid >> 1, wn = wid & 1;
  const int row0 = blockIdx.x * BM, col0 = blockIdx.y * BN;
  v8f acc[2][4] = {};

  auto stage = [&](int buf, int k0) {
#pragma unroll
    for (int i = 0; i < 2; ++i) {                    // A: 512 b128 slots
      int slot = tid + i * 256;
      int r = slot >> 2, c8 = slot & 3;
      async_load_b128(&As[buf][r * LDA + c8 * 8],
                      X + (size_t)(row0 + r) * NC1 + k0 + c8 * 8);
    }
#pragma unroll
    for (int i = 0; i < 2; ++i) {                    // B: 512 b128 slots
      int slot = tid + i * 256;
      int r = slot >> 4, c8 = slot & 15;
      async_load_b128(&Bs[buf][r * LDB + c8 * 8],
                      W2 + (size_t)(k0 + r) * FDIM + col0 + c8 * 8);
    }
  };

  stage(0, 0);
  wait_asynccnt0();
  __syncthreads();

  int cur = 0;
  for (int k0 = 0; k0 < NC1; k0 += BK, cur ^= 1) {
    const int nxt = cur ^ 1;
    if (k0 + BK < NC1) stage(nxt, k0 + BK);
    v16bf a0 = load_a_frag(&As[cur][(wm * 32) * LDA], LDA, lane);
    v16bf a1 = load_a_frag(&As[cur][(wm * 32 + 16) * LDA], LDA, lane);
#pragma unroll
    for (int j = 0; j < 4; ++j) {
      v16bf b = load_b_frag(&Bs[cur][0], LDB, wn * 64 + j * 16, lane);
      acc[0][j] = wmma_bf16(a0, b, acc[0][j]);
      acc[1][j] = wmma_bf16(a1, b, acc[1][j]);
    }
    wait_asynccnt0();
    __syncthreads();
  }
  const int h = lane >> 4, cn = lane & 15;
#pragma unroll
  for (int i = 0; i < 2; ++i)
#pragma unroll
    for (int j = 0; j < 4; ++j) {
      int col = col0 + wn * 64 + j * 16 + cn;
      float d = Dvec[col];
#pragma unroll
      for (int r = 0; r < 8; ++r) {
        size_t row = (size_t)row0 + wm * 32 + i * 16 + h * 8 + r;
        size_t idx = row * FDIM + col;
        y[idx] = acc[i][j][r] + d * u[idx];
      }
    }
}

// ---------------------------------------------------------------------------
extern "C" void kernel_launch(void* const* d_in, const int* in_sizes, int n_in,
                              void* d_out, int out_size, void* d_ws, size_t ws_size,
                              hipStream_t stream) {
  const float* u          = (const float*)d_in[0];
  const float* log_A_real = (const float*)d_in[1];
  const float* A_imag     = (const float*)d_in[2];
  const float* B_re       = (const float*)d_in[3];
  const float* B_im       = (const float*)d_in[4];
  const float* C_re       = (const float*)d_in[5];
  const float* C_im       = (const float*)d_in[6];
  const float* Dvec       = (const float*)d_in[7];
  const float* log_dt     = (const float*)d_in[8];
  float* y = (float*)d_out;

  char* ws = (char*)d_ws;
  auto carve = [&](size_t bytes) {
    char* p = ws;
    ws += (bytes + 255) & ~(size_t)255;
    return p;
  };
  __bf16* W1    = (__bf16*)carve((size_t)FDIM * NC1 * sizeof(__bf16));
  __bf16* W2    = (__bf16*)carve((size_t)NC1 * FDIM * sizeof(__bf16));
  float2* A_bar = (float2*)carve((size_t)NDIM * sizeof(float2));
  float2* Apow  = (float2*)carve((size_t)NDIM * (LCHUNK + 1) * sizeof(float2));
  float2* carry = (float2*)carve((size_t)BDIM * NCHUNK * NDIM * sizeof(float2));
  float2* hin   = (float2*)carve((size_t)BDIM * NCHUNK * NDIM * sizeof(float2));
  float*  Bu    = (float*)carve((size_t)MROWS * NC1 * sizeof(float));
  __bf16* X     = (__bf16*)carve((size_t)MROWS * NC1 * sizeof(__bf16));

  prep_params_kernel<<<1, 128, 0, stream>>>(log_A_real, A_imag, log_dt, A_bar, Apow);
  prep_w1_kernel<<<(NDIM * FDIM) / 256, 256, 0, stream>>>(B_re, B_im, log_A_real, A_imag, A_bar, W1);
  prep_w2_kernel<<<(NDIM * FDIM) / 256, 256, 0, stream>>>(C_re, C_im, W2);
  gemm1_kernel<<<dim3(MROWS / 128, NC1 / 64), 256, 0, stream>>>(u, W1, Bu);
  scan_chunk_kernel<<<dim3(NCHUNK, BDIM), 256, 0, stream>>>(Bu, A_bar, carry);
  carry_scan_kernel<<<BDIM, NDIM, 0, stream>>>(carry, Apow, hin);
  fixup_kernel<<<dim3(NCHUNK, BDIM), 256, 0, stream>>>(Bu, hin, Apow, X);
  gemm2_kernel<<<dim3(MROWS / 128, FDIM / 128), 256, 0, stream>>>(X, W2, u, Dvec, y);
}
